// MultiViewMatching_14972255993871
// MI455X (gfx1250) — matrix operand order repaired and checked by reading, not compiled, not run on hardware
//
#include <hip/hip_runtime.h>
#include <hip/hip_bf16.h>
#include <math.h>

// ---------------------------------------------------------------------------
// MultiViewMatching for MI455X (gfx1250, wave32).
//   d_in[0] = imgs  [4096 x 512] f32
//   d_in[1] = caps  [4096 x 512] f32
//   d_out   = scores [4096*4096] f32  ++  scores_ot [1] f32
// ---------------------------------------------------------------------------

#define NDIM 4096
#define KDIM 512

typedef __attribute__((ext_vector_type(16))) __bf16 v16bf;
typedef __attribute__((ext_vector_type(8)))  __bf16 v8bf;
typedef __attribute__((ext_vector_type(8)))  float  v8f;

union bfrag {
  v16bf v;
  v8bf  h[2];
};

__device__ inline void split_bf16_16(const float* v, v16bf& hi, v16bf& lo) {
#pragma unroll
  for (int e = 0; e < 16; ++e) {
    float x = v[e];
    __bf16 h = (__bf16)x;           // RNE to bf16 (high part)
    hi[e] = h;
    lo[e] = (__bf16)(x - (float)h); // residual (low part)
  }
}

// ---------------------------------------------------------------------------
// Row L2-norms -> inverse norms 1/(||x||+1e-12). blocks 0..4095 -> imgs,
// blocks 4096..8191 -> caps.
// ---------------------------------------------------------------------------
__global__ __launch_bounds__(256) void norms_kernel(
    const float* __restrict__ imgs, const float* __restrict__ caps,
    float* __restrict__ invI, float* __restrict__ invC) {
  const int r = blockIdx.x;
  const float* src = (r < NDIM) ? (imgs + (size_t)r * KDIM)
                                : (caps + (size_t)(r - NDIM) * KDIM);
  float acc = 0.0f;
  for (int k = threadIdx.x; k < KDIM; k += 256) {
    float x = src[k];
    acc += x * x;
  }
  __shared__ float sh[256];
  sh[threadIdx.x] = acc;
  __syncthreads();
  for (int s = 128; s > 0; s >>= 1) {
    if (threadIdx.x < s) sh[threadIdx.x] += sh[threadIdx.x + s];
    __syncthreads();
  }
  if (threadIdx.x == 0) {
    float v = 1.0f / (sqrtf(sh[0]) + 1e-12f);
    if (r < NDIM) invI[r] = v; else invC[r - NDIM] = v;
  }
}

// ---------------------------------------------------------------------------
// Split-bf16 WMMA GEMM: scores = imgs @ caps^T  (f32-grade accuracy via
// Ahi*Bhi + Ahi*Blo + Alo*Bhi on v_wmma_f32_16x16x32_bf16).
//
// Block = 256 threads = 8 waves; tile 128(M) x 64(N); each wave owns a 16x64
// strip (4 accumulators).  The 64x32 B panel is loaded + split to bf16 hi/lo
// ONCE per block per k-step and staged in LDS as [n][kk], kk in natural order
// (pitch 40 bf16 to avoid bank conflicts), so each wave only does ds_load
// for its B fragments -> 8x less global traffic & conversion VALU.
// Epilogue fuses cosine distance + per-block min/max for the threshold.
// ---------------------------------------------------------------------------
__global__ __launch_bounds__(256) void gemm_scores_kernel(
    const float* __restrict__ imgs, const float* __restrict__ caps,
    float* __restrict__ scores, const float* __restrict__ invI,
    const float* __restrict__ invC, float* __restrict__ bmin,
    float* __restrict__ bmax) {
  const int tid  = threadIdx.x;
  const int lane = tid & 31;
  const int wave = tid >> 5;     // 0..7
  const int half = lane >> 4;    // 0 | 1
  const int l16  = lane & 15;

  const int rowBase = blockIdx.y * 128 + wave * 16;
  const int colBase = blockIdx.x * 64;

  // B panel staged as bf16 hi/lo: [n=0..63][kk=0..31], pitch 40 (pad 8)
  __shared__ __align__(16) __bf16 ldsHi[64 * 40];
  __shared__ __align__(16) __bf16 ldsLo[64 * 40];

  const float* arow = imgs + (size_t)(rowBase + l16) * KDIM;

  v8f acc[4] = {};

  for (int k0 = 0; k0 < KDIM; k0 += 32) {
    // ---- Stage B panel (cooperative, once per block) -------------------
    __syncthreads();  // previous compute finished reading LDS
    {
      const int n  = tid >> 2;       // 0..63  (panel column = colBase + n)
      const int kb = (tid & 3) * 8;  // 0,8,16,24
      const float4* p =
          (const float4*)(caps + (size_t)(colBase + n) * KDIM + k0 + kb);
      float4 x0 = p[0], x1 = p[1];
      float xv[8] = {x0.x, x0.y, x0.z, x0.w, x1.x, x1.y, x1.z, x1.w};
      v8bf h8, l8;
#pragma unroll
      for (int e = 0; e < 8; ++e) {
        float x = xv[e];
        __bf16 h = (__bf16)x;
        h8[e] = h;
        l8[e] = (__bf16)(x - (float)h);
      }
      *(v8bf*)(&ldsHi[n * 40 + kb]) = h8;
      *(v8bf*)(&ldsLo[n * 40 + kb]) = l8;
    }
    __syncthreads();  // panel visible to all waves

    // ---- A fragment (16x32 bf16 layout, ISA 7.12.2):
    // lanes 0-15: K = {k0+0..7, k0+16..23}; lanes 16-31: K = {+8..15, +24..31}
    float av[16];
    {
      const int b1 = k0 + (half ? 8 : 0);
      const int b2 = k0 + 16 + (half ? 8 : 0);
      const float4* p = (const float4*)arow;
      float4 x0 = p[b1 >> 2], x1 = p[(b1 >> 2) + 1];
      float4 x2 = p[b2 >> 2], x3 = p[(b2 >> 2) + 1];
      av[0]=x0.x; av[1]=x0.y; av[2]=x0.z; av[3]=x0.w;
      av[4]=x1.x; av[5]=x1.y; av[6]=x1.z; av[7]=x1.w;
      av[8]=x2.x; av[9]=x2.y; av[10]=x2.z; av[11]=x2.w;
      av[12]=x3.x; av[13]=x3.y; av[14]=x3.z; av[15]=x3.w;
    }
    v16bf ahi, alo;
    split_bf16_16(av, ahi, alo);

    // ---- 4 N-tiles: B fragments from LDS (element e <-> kk = half*8 + e for
    // e<8, 16 + half*8 + (e-8) for e>=8; n = t*16 + l16), then the 3-term
    // split-bf16 WMMA accumulation.
#pragma unroll
    for (int t = 0; t < 4; ++t) {
      const int nloc = t * 16 + l16;
      const __bf16* hp = &ldsHi[nloc * 40 + half * 8];
      const __bf16* lp = &ldsLo[nloc * 40 + half * 8];
      bfrag bh, bl;
      bh.h[0] = *(const v8bf*)(hp);
      bh.h[1] = *(const v8bf*)(hp + 16);
      bl.h[0] = *(const v8bf*)(lp);
      bl.h[1] = *(const v8bf*)(lp + 16);

      acc[t] = __builtin_amdgcn_wmma_f32_16x16x32_bf16(
          false, ahi, false, bh.v, (short)0, acc[t], false, false);
      acc[t] = __builtin_amdgcn_wmma_f32_16x16x32_bf16(
          false, ahi, false, bl.v, (short)0, acc[t], false, false);
      acc[t] = __builtin_amdgcn_wmma_f32_16x16x32_bf16(
          false, alo, false, bh.v, (short)0, acc[t], false, false);
    }
  }

  // ---- Epilogue: store scores, fuse cosine distance, track block min/max.
  float invI_row[8];
#pragma unroll
  for (int v = 0; v < 8; ++v) invI_row[v] = invI[rowBase + v + 8 * half];

  float mn = 3.4e38f, mx = -3.4e38f;
#pragma unroll
  for (int t = 0; t < 4; ++t) {
    const int col = colBase + t * 16 + l16;
    const float ic = invC[col];
#pragma unroll
    for (int v = 0; v < 8; ++v) {
      const int row = rowBase + v + 8 * half;
      float s = acc[t][v];
      scores[(size_t)row * NDIM + col] = s;
      float cosd = 1.0f - s * invI_row[v] * ic;
      mn = fminf(mn, cosd);
      mx = fmaxf(mx, cosd);
    }
  }

  __shared__ float smn[256], smx[256];
  smn[tid] = mn; smx[tid] = mx;
  __syncthreads();
  for (int s = 128; s > 0; s >>= 1) {
    if (tid < s) {
      smn[tid] = fminf(smn[tid], smn[tid + s]);
      smx[tid] = fmaxf(smx[tid], smx[tid + s]);
    }
    __syncthreads();
  }
  if (tid == 0) {
    const int bid = blockIdx.y * gridDim.x + blockIdx.x;
    bmin[bid] = smn[0];
    bmax[bid] = smx[0];
  }
}

// ---------------------------------------------------------------------------
// Reduce 2048 block min/max partials -> thr = min + 0.1*(max-min)
// ---------------------------------------------------------------------------
__global__ __launch_bounds__(256) void thr_reduce_kernel(
    const float* __restrict__ bmin, const float* __restrict__ bmax,
    float* __restrict__ thr_p) {
  float mn = 3.4e38f, mx = -3.4e38f;
  for (int i = threadIdx.x; i < 2048; i += 256) {
    mn = fminf(mn, bmin[i]);
    mx = fmaxf(mx, bmax[i]);
  }
  __shared__ float smn[256], smx[256];
  smn[threadIdx.x] = mn; smx[threadIdx.x] = mx;
  __syncthreads();
  for (int s = 128; s > 0; s >>= 1) {
    if (threadIdx.x < s) {
      smn[threadIdx.x] = fminf(smn[threadIdx.x], smn[threadIdx.x + s]);
      smx[threadIdx.x] = fmaxf(smx[threadIdx.x], smx[threadIdx.x + s]);
    }
    __syncthreads();
  }
  if (threadIdx.x == 0) thr_p[0] = smn[0] + 0.1f * (smx[0] - smn[0]);
}

__global__ __launch_bounds__(256) void ipot_init_kernel(
    float* __restrict__ S, float* __restrict__ sigma) {
  int j = blockIdx.x * 256 + threadIdx.x;
  S[j] = 1.0f;
  sigma[j] = 1.0f / (float)NDIM;
}

// ---------------------------------------------------------------------------
// IPOT row pass (iteration k): r_i = sum_j E_ij * S_j * sigma_j,
// then D_i <- 1/(bs*r_i)   (uses T_k = E_k o D S^T structure; E recomputed
// from `scores`, which stays resident in the 192MB L2).
// ---------------------------------------------------------------------------
__global__ __launch_bounds__(256) void ipot_row_kernel(
    const float* __restrict__ scores, const float* __restrict__ invI,
    const float* __restrict__ invC, const float* __restrict__ thr_p,
    const float* __restrict__ S, const float* __restrict__ sigma,
    float* __restrict__ D, float mk) {
  const int i = blockIdx.x;
  const float ii = invI[i];
  const float thr = thr_p[0];
  const float* row = scores + (size_t)i * NDIM;
  float acc = 0.0f;
  for (int j = threadIdx.x; j < NDIM; j += 256) {
    float cosd = 1.0f - row[j] * ii * invC[j];
    float Cs = fmaxf(cosd - thr, 0.0f);
    acc += expf(mk * Cs) * S[j] * sigma[j];
  }
  __shared__ float sh[256];
  sh[threadIdx.x] = acc;
  __syncthreads();
  for (int s = 128; s > 0; s >>= 1) {
    if (threadIdx.x < s) sh[threadIdx.x] += sh[threadIdx.x + s];
    __syncthreads();
  }
  if (threadIdx.x == 0) D[i] = 1.0f / ((float)NDIM * sh[0]);
}

// Column pass stage 1: partial c_j over 128-row chunks (deterministic,
// coalesced; no float atomics).
__global__ __launch_bounds__(256) void ipot_col1_kernel(
    const float* __restrict__ scores, const float* __restrict__ invI,
    const float* __restrict__ invC, const float* __restrict__ thr_p,
    const float* __restrict__ D, float* __restrict__ cpart, float mk) {
  const int j = blockIdx.x * 256 + threadIdx.x;
  const int i0 = blockIdx.y * 128;
  const float thr = thr_p[0];
  const float ic = invC[j];
  float acc = 0.0f;
  for (int i = i0; i < i0 + 128; ++i) {
    float cosd = 1.0f - scores[(size_t)i * NDIM + j] * invI[i] * ic;
    float Cs = fmaxf(cosd - thr, 0.0f);
    acc += expf(mk * Cs) * D[i];
  }
  cpart[(size_t)blockIdx.y * NDIM + j] = acc;
}

// Column pass stage 2: c_j = sum of 32 partials; sigma_j = 1/(bt*S_j*c_j);
// S_j <- 1/(bt*c_j).
__global__ __launch_bounds__(256) void ipot_col2_kernel(
    const float* __restrict__ cpart, float* __restrict__ S,
    float* __restrict__ sigma) {
  const int j = blockIdx.x * 256 + threadIdx.x;
  float c = 0.0f;
  for (int rc = 0; rc < 32; ++rc) c += cpart[(size_t)rc * NDIM + j];
  float Sold = S[j];
  sigma[j] = 1.0f / ((float)NDIM * Sold * c);
  S[j] = 1.0f / ((float)NDIM * c);
}

// Final: rowsum_i = D_i * sum_j Cs_ij * E5_ij * S_j
__global__ __launch_bounds__(256) void final_row_kernel(
    const float* __restrict__ scores, const float* __restrict__ invI,
    const float* __restrict__ invC, const float* __restrict__ thr_p,
    const float* __restrict__ D, const float* __restrict__ S,
    float* __restrict__ rowsum) {
  const int i = blockIdx.x;
  const float ii = invI[i];
  const float thr = thr_p[0];
  const float* row = scores + (size_t)i * NDIM;
  float acc = 0.0f;
  for (int j = threadIdx.x; j < NDIM; j += 256) {
    float cosd = 1.0f - row[j] * ii * invC[j];
    float Cs = fmaxf(cosd - thr, 0.0f);
    acc += Cs * expf(-10.0f * Cs) * S[j];
  }
  __shared__ float sh[256];
  sh[threadIdx.x] = acc;
  __syncthreads();
  for (int s = 128; s > 0; s >>= 1) {
    if (threadIdx.x < s) sh[threadIdx.x] += sh[threadIdx.x + s];
    __syncthreads();
  }
  if (threadIdx.x == 0) rowsum[i] = D[i] * sh[0];
}

__global__ __launch_bounds__(256) void final_reduce_kernel(
    const float* __restrict__ rowsum, float* __restrict__ out) {
  float acc = 0.0f;
  for (int i = threadIdx.x; i < NDIM; i += 256) acc += rowsum[i];
  __shared__ float sh[256];
  sh[threadIdx.x] = acc;
  __syncthreads();
  for (int s = 128; s > 0; s >>= 1) {
    if (threadIdx.x < s) sh[threadIdx.x] += sh[threadIdx.x + s];
    __syncthreads();
  }
  if (threadIdx.x == 0) out[0] = sh[0];
}

// ---------------------------------------------------------------------------
extern "C" void kernel_launch(void* const* d_in, const int* in_sizes, int n_in,
                              void* d_out, int out_size, void* d_ws,
                              size_t ws_size, hipStream_t stream) {
  (void)in_sizes; (void)n_in; (void)out_size; (void)ws_size;
  const float* imgs = (const float*)d_in[0];
  const float* caps = (const float*)d_in[1];
  float* scores = (float*)d_out;                       // [4096*4096]
  float* out_ot = (float*)d_out + (size_t)NDIM * NDIM; // scalar

  // Workspace layout (floats)
  float* w = (float*)d_ws;
  float* invI   = w;                 // 4096
  float* invC   = w + 4096;          // 4096
  float* bmin   = w + 8192;          // 2048
  float* bmax   = w + 10240;         // 2048
  float* thr_p  = w + 12288;         // 1 (padded to 12352)
  float* D      = w + 12352;         // 4096
  float* S      = w + 16448;         // 4096
  float* sigma  = w + 20544;         // 4096
  float* cpart  = w + 24640;         // 32*4096 = 131072
  float* rowsum = w + 155712;        // 4096

  // 1) inverse row norms
  norms_kernel<<<2 * NDIM, 256, 0, stream>>>(imgs, caps, invI, invC);

  // 2) WMMA GEMM (scores) + fused cosine min/max partials
  gemm_scores_kernel<<<dim3(NDIM / 64, NDIM / 128), 256, 0, stream>>>(
      imgs, caps, scores, invI, invC, bmin, bmax);

  // 3) threshold
  thr_reduce_kernel<<<1, 256, 0, stream>>>(bmin, bmax, thr_p);

  // 4) IPOT (rank-1 factored; T never materialized)
  ipot_init_kernel<<<NDIM / 256, 256, 0, stream>>>(S, sigma);
  for (int k = 1; k <= 5; ++k) {
    const float mk = -2.0f * (float)k;  // exp(-k*Cs/0.5)
    ipot_row_kernel<<<NDIM, 256, 0, stream>>>(scores, invI, invC, thr_p, S,
                                              sigma, D, mk);
    ipot_col1_kernel<<<dim3(NDIM / 256, 32), 256, 0, stream>>>(
        scores, invI, invC, thr_p, D, cpart, mk);
    ipot_col2_kernel<<<NDIM / 256, 256, 0, stream>>>(cpart, S, sigma);
  }

  // 5) scores_ot = sum(Cs * T5)
  final_row_kernel<<<NDIM, 256, 0, stream>>>(scores, invI, invC, thr_p, D, S,
                                             rowsum);
  final_reduce_kernel<<<1, 256, 0, stream>>>(rowsum, out_ot);
}
